// InEKFTracker_23622320128495
// MI455X (gfx1250) — matrix operand order repaired and checked by reading, not compiled, not run on hardware
//
#include <hip/hip_runtime.h>
#include <hip/hip_bf16.h>
#include <math.h>

typedef _Float16 v16h __attribute__((ext_vector_type(16)));
typedef _Float16 v8h  __attribute__((ext_vector_type(8)));
typedef float    v8f  __attribute__((ext_vector_type(8)));
typedef float    v4f  __attribute__((ext_vector_type(4)));

#define MTOT 8192   // B*T rows
#define DK   1024   // content dim
#define HIDN 64     // MLP hidden
#define NOUT 1024   // H*NB
#define TLEN 2048

// ---------------------------------------------------------------------------
// Prep: W1 (DK x HIDN f32) -> W1T (HIDN x DK f16)   [B-frags for GEMM1]
//       W2 (HIDN x NOUT f32) -> W2T (NOUT x HIDN f16) [B-frags for GEMM2]
// ---------------------------------------------------------------------------
__global__ void prep_kernel(const float* __restrict__ W1, const float* __restrict__ W2,
                            _Float16* __restrict__ W1T, _Float16* __restrict__ W2T) {
    int idx = blockIdx.x * blockDim.x + threadIdx.x;   // 0 .. 131071
    if (idx < HIDN * DK) {
        int n = idx / DK, k = idx % DK;                // write coalesced
        W1T[idx] = (_Float16)W1[k * HIDN + n];
    } else {
        int j = idx - HIDN * DK;
        int n = j / HIDN, k = j % HIDN;
        W2T[j] = (_Float16)W2[k * NOUT + n];
    }
}

// ---------------------------------------------------------------------------
// Fused gate MLP: gates = sigmoid(gelu(X@W1 + b1)@W2 + b2)
// 128 threads = 4 waves; each wave owns a 16-row M tile (block: 64 rows).
// GEMM loops are software-pipelined: next K-step fragments load while the
// current step's WMMAs execute (wait <= prefetch depth instead of 0).
// ---------------------------------------------------------------------------
__global__ __launch_bounds__(128) void mlp_kernel(
    const float* __restrict__ X, const _Float16* __restrict__ W1T,
    const float* __restrict__ b1, const _Float16* __restrict__ W2T,
    const float* __restrict__ b2, float* __restrict__ gates) {

    __shared__ __attribute__((aligned(16))) _Float16 hstage[4][16][72]; // 72-half stride: conflict-free

    const int wave  = threadIdx.x >> 5;
    const int lane  = threadIdx.x & 31;
    const int lrow  = lane & 15;     // row (A frag) or column (B/C frag) index
    const int khalf = lane >> 4;     // which K-half this lane holds

    // ---------------- GEMM1: (16 x 1024) @ (1024 x 64) ----------------
    const int row = blockIdx.x * 64 + wave * 16 + lrow;   // global M row for A frag
    const float* xrow = X + (size_t)row * DK;
    const _Float16* wb0 = W1T + (size_t)(0 * 16 + lrow) * DK + khalf * 16;
    const _Float16* wb1 = W1T + (size_t)(1 * 16 + lrow) * DK + khalf * 16;
    const _Float16* wb2 = W1T + (size_t)(2 * 16 + lrow) * DK + khalf * 16;
    const _Float16* wb3 = W1T + (size_t)(3 * 16 + lrow) * DK + khalf * 16;

    auto loadA = [&](int kb) -> v16h {
        v4f a0 = *(const v4f*)(xrow + kb + khalf * 16);
        v4f a1 = *(const v4f*)(xrow + kb + khalf * 16 + 4);
        v4f a2 = *(const v4f*)(xrow + kb + khalf * 16 + 8);
        v4f a3 = *(const v4f*)(xrow + kb + khalf * 16 + 12);
        v16h af;
#pragma unroll
        for (int i = 0; i < 4; ++i) {
            af[i]      = (_Float16)a0[i];
            af[4 + i]  = (_Float16)a1[i];
            af[8 + i]  = (_Float16)a2[i];
            af[12 + i] = (_Float16)a3[i];
        }
        return af;
    };

    v8f acc0 = {}, acc1 = {}, acc2 = {}, acc3 = {};
    v16h af = loadA(0);
    v16h b0 = *(const v16h*)(wb0), b1f = *(const v16h*)(wb1),
         b2f = *(const v16h*)(wb2), b3f = *(const v16h*)(wb3);

    for (int ks = 0; ks < 32; ++ks) {
        const v16h caf = af, cb0 = b0, cb1 = b1f, cb2 = b2f, cb3 = b3f;
        if (ks < 31) {                       // prefetch next K-step while WMMAs run
            const int kb = (ks + 1) * 32;
            af  = loadA(kb);
            b0  = *(const v16h*)(wb0 + kb);
            b1f = *(const v16h*)(wb1 + kb);
            b2f = *(const v16h*)(wb2 + kb);
            b3f = *(const v16h*)(wb3 + kb);
        }
        acc0 = __builtin_amdgcn_wmma_f32_16x16x32_f16(false, caf, false, cb0, (short)0, acc0, false, false);
        acc1 = __builtin_amdgcn_wmma_f32_16x16x32_f16(false, caf, false, cb1, (short)0, acc1, false, false);
        acc2 = __builtin_amdgcn_wmma_f32_16x16x32_f16(false, caf, false, cb2, (short)0, acc2, false, false);
        acc3 = __builtin_amdgcn_wmma_f32_16x16x32_f16(false, caf, false, cb3, (short)0, acc3, false, false);
    }

    // Bias + exact GELU, stage to LDS as f16 (C layout: vgpr i -> row khalf*8+i, col = nt*16+lrow)
    {
        v8f* accs[4] = {&acc0, &acc1, &acc2, &acc3};
#pragma unroll
        for (int nt = 0; nt < 4; ++nt) {
            const float bias = b1[nt * 16 + lrow];
#pragma unroll
            for (int i = 0; i < 8; ++i) {
                float x = (*accs[nt])[i] + bias;
                float g = 0.5f * x * (1.0f + erff(x * 0.70710678118654752f));
                hstage[wave][khalf * 8 + i][nt * 16 + lrow] = (_Float16)g;
            }
        }
    }
    __syncthreads();

    // Re-read hidden tile in A-fragment layout (lane = row lrow, K chunks per khalf)
    v16h ah0, ah1;
    {
        v8h c0 = *(const v8h*)&hstage[wave][lrow][0 + khalf * 16];
        v8h c1 = *(const v8h*)&hstage[wave][lrow][8 + khalf * 16];
        ah0 = __builtin_shufflevector(c0, c1, 0,1,2,3,4,5,6,7,8,9,10,11,12,13,14,15);
        v8h c2 = *(const v8h*)&hstage[wave][lrow][32 + khalf * 16];
        v8h c3 = *(const v8h*)&hstage[wave][lrow][40 + khalf * 16];
        ah1 = __builtin_shufflevector(c2, c3, 0,1,2,3,4,5,6,7,8,9,10,11,12,13,14,15);
    }

    // ---------------- GEMM2: (16 x 64) @ (64 x 1024), pipelined over 64 N tiles ----------------
    const int gmbase = blockIdx.x * 64 + wave * 16 + khalf * 8;  // output row base for this lane's C frag
    const _Float16* w2p = W2T + (size_t)lrow * HIDN + khalf * 16;
    v16h nb0 = *(const v16h*)(w2p);
    v16h nb1 = *(const v16h*)(w2p + 32);
    for (int nt2 = 0; nt2 < 64; ++nt2) {
        const v16h bf0 = nb0, bf1 = nb1;
        if (nt2 < 63) {                      // prefetch next tile's B frags
            w2p += (size_t)16 * HIDN;
            nb0 = *(const v16h*)(w2p);
            nb1 = *(const v16h*)(w2p + 32);
        }
        v8f c = {};
        c = __builtin_amdgcn_wmma_f32_16x16x32_f16(false, ah0, false, bf0, (short)0, c, false, false);
        c = __builtin_amdgcn_wmma_f32_16x16x32_f16(false, ah1, false, bf1, (short)0, c, false, false);
        const float bias2 = b2[nt2 * 16 + lrow];
        const int coln = nt2 * 16 + lrow;
#pragma unroll
        for (int i = 0; i < 8; ++i) {
            float x = c[i] + bias2;
            float g = 1.0f / (1.0f + expf(-x));
            gates[(size_t)(gmbase + i) * NOUT + coln] = g;   // keep RT: scan re-reads from L2
        }
    }
}

// ---------------------------------------------------------------------------
// Sequential Kalman-variance scan: one thread per (b,h,nb) channel.
// lane = nb -> fully coalesced. Depth-8 register prefetch ring (16 outstanding
// loads/thread) + global_prefetch 4 groups ahead; sigma stores non-temporal
// so the write-once output does not evict delta/gates from L2.
// ---------------------------------------------------------------------------
__global__ __launch_bounds__(256) void scan_kernel(
    const float* __restrict__ delta, const float* __restrict__ gates,
    const float* __restrict__ omega, const float* __restrict__ logQ,
    const float* __restrict__ logR, const float* __restrict__ logS0,
    float* __restrict__ sigma_out) {

    const int tid = blockIdx.x * 256 + threadIdx.x;  // 0..4095
    const int ch  = tid & 1023;                      // h*64 + nb
    const int b   = tid >> 10;

    const float om = omega[ch];
    const float Q  = expf(logQ[ch]);
    const float R  = expf(logR[ch]);
    float sigma    = expf(logS0[ch]);

    const size_t base = (size_t)b * TLEN * 1024 + ch;

    constexpr int PF = 8;
    float dbuf[PF], gbuf[PF];
#pragma unroll
    for (int j = 0; j < PF; ++j) {
        dbuf[j] = delta[base + (size_t)j * 1024];
        gbuf[j] = gates[base + (size_t)j * 1024];
    }

    for (int t = 0; t < TLEN; t += PF) {
        {   // long-range prefetch into cache hierarchy (global_prefetch_b8)
            int tpf = t + 4 * PF;
            if (tpf > TLEN - 1) tpf = TLEN - 1;
            __builtin_prefetch(&delta[base + (size_t)tpf * 1024], 0, 0);
            __builtin_prefetch(&gates[base + (size_t)tpf * 1024], 0, 0);
        }
#pragma unroll
        for (int j = 0; j < PF; ++j) {
            const float d = dbuf[j];
            const float g = gbuf[j];
            const int tp = t + j + PF;
            if (tp < TLEN) {                 // refill ring slot (uniform branch)
                dbuf[j] = delta[base + (size_t)tp * 1024];
                gbuf[j] = gates[base + (size_t)tp * 1024];
            }
            float th = d * om;
            th = th * th;
            sigma = sigma + Q * th;
            const float K = sigma / (sigma + R);
            sigma = sigma * (1.0f - g * K);
            __builtin_nontemporal_store(sigma, &sigma_out[base + (size_t)(t + j) * 1024]);
        }
    }
}

// ---------------------------------------------------------------------------
extern "C" void kernel_launch(void* const* d_in, const int* in_sizes, int n_in,
                              void* d_out, int out_size, void* d_ws, size_t ws_size,
                              hipStream_t stream) {
    const float* delta   = (const float*)d_in[0];
    const float* content = (const float*)d_in[1];
    const float* omega   = (const float*)d_in[2];
    const float* logQ    = (const float*)d_in[3];
    const float* logR    = (const float*)d_in[4];
    const float* logS0   = (const float*)d_in[5];
    const float* W1      = (const float*)d_in[6];
    const float* b1      = (const float*)d_in[7];
    const float* W2      = (const float*)d_in[8];
    const float* b2      = (const float*)d_in[9];

    float* sigma_out = (float*)d_out;
    float* gates_out = sigma_out + (size_t)MTOT * NOUT;  // tuple: (sigma, gates)

    _Float16* W1T = (_Float16*)d_ws;            // 64 x 1024 f16 (128 KB)
    _Float16* W2T = W1T + (size_t)HIDN * DK;    // 1024 x 64 f16 (128 KB)

    prep_kernel<<<(HIDN * DK + NOUT * HIDN) / 256, 256, 0, stream>>>(W1, W2, W1T, W2T);
    mlp_kernel<<<MTOT / 64, 128, 0, stream>>>(content, W1T, b1, W2T, b2, gates_out);
    scan_kernel<<<4096 / 256, 256, 0, stream>>>(delta, gates_out, omega, logQ, logR, logS0, sigma_out);
}
